// SelfAttention_41120016892354
// MI455X (gfx1250) — matrix operand order repaired and checked by reading, not compiled, not run on hardware
//
#include <hip/hip_runtime.h>
#include <hip/hip_bf16.h>

// ---------- types ----------
typedef __bf16 bf16_t;
typedef bf16_t v16bf __attribute__((ext_vector_type(16)));
typedef float  v8f   __attribute__((ext_vector_type(8)));

__device__ __forceinline__ v8f wmma_bf16(v16bf a, v16bf b, v8f c) {
  // D = A(16x32 bf16) * B(32x16 bf16) + C(16x16 f32)
  return __builtin_amdgcn_wmma_f32_16x16x32_bf16(false, a, false, b, (short)0, c, false, false);
}

__device__ __forceinline__ v8f v8f_zero() {
  v8f z;
#pragma unroll
  for (int i = 0; i < 8; ++i) z[i] = 0.0f;
  return z;
}

// load 16 contiguous bf16 (32B) from global/LDS
__device__ __forceinline__ v16bf load16(const bf16_t* p) {
  v16bf r;
  uint4* d = reinterpret_cast<uint4*>(&r);
  d[0] = *reinterpret_cast<const uint4*>(p);
  d[1] = *reinterpret_cast<const uint4*>(p + 8);
  return r;
}

// load A-operand as two 8-element (16B) chunks (A layout: per-lane two strided 8-chunks)
__device__ __forceinline__ v16bf load8x2(const bf16_t* p0, const bf16_t* p1) {
  v16bf r;
  uint4* d = reinterpret_cast<uint4*>(&r);
  d[0] = *reinterpret_cast<const uint4*>(p0);
  d[1] = *reinterpret_cast<const uint4*>(p1);
  return r;
}

// ---------- ws layout (bytes) ----------
#define OFF_WCAT   0u            // bf16 [192][256]   = 98304
#define OFF_WOT    98304u        // bf16 [256][128]   = 65536
#define OFF_THETA  163840u       // bf16 [65536][32]  = 4 MB
#define OFF_PHIF   4358144u      // bf16 [65536][32]  = 4 MB
#define OFF_GF     8552448u      // bf16 [65536][128] = 16 MB
#define OFF_PHIP   25329664u     // bf16 [16][1024][32] = 1 MB
#define OFF_GT     26378240u     // bf16 [16][128][1024] = 4 MB

// ============================================================
// Kernel 0: weight prep -> bf16, transposed (cout-major rows)
// ============================================================
__global__ __launch_bounds__(256) void prep_w(
    const float* __restrict__ Wt, const float* __restrict__ Wp,
    const float* __restrict__ Wg, const float* __restrict__ Wo,
    bf16_t* __restrict__ WcatT, bf16_t* __restrict__ WoT) {
  int id = blockIdx.x * 256 + threadIdx.x;
  if (id < 192 * 256) {
    int cin = id & 255, cout = id >> 8;
    float v;
    if (cout < 32)       v = Wt[cin * 32 + cout];
    else if (cout < 64)  v = Wp[cin * 32 + (cout - 32)];
    else                 v = Wg[cin * 128 + (cout - 64)];
    WcatT[cout * 256 + cin] = (bf16_t)v;
  } else {
    int j = id - 192 * 256;
    if (j < 256 * 128) {
      int vch = j & 127, cout = j >> 7;
      WoT[cout * 128 + vch] = (bf16_t)Wo[vch * 256 + cout];  // Wo is [128][256]
    }
  }
}

// ============================================================
// Kernel 1: fused 1x1 convs (GEMM [65536,256] x [256,192])
// 8 waves * 16 rows = 128 rows/WG; 12 N-tiles; 8 K-steps.
// B tiles loaded in groups of 4 so the load clause overlaps WMMAs.
// ============================================================
__global__ __launch_bounds__(256) void conv_qkv(
    const float* __restrict__ x, const bf16_t* __restrict__ WcatT,
    const float* __restrict__ bth, const float* __restrict__ bph,
    const float* __restrict__ bg,
    bf16_t* __restrict__ theta, bf16_t* __restrict__ phiF, bf16_t* __restrict__ gF) {
  const int lane = threadIdx.x & 31;
  const int wave = threadIdx.x >> 5;
  const int half = lane >> 4;
  const int l16  = lane & 15;
  const long qb = (long)blockIdx.x * 128 + wave * 16;
  const long q  = qb + l16;

  v8f acc[12];
#pragma unroll
  for (int t = 0; t < 12; ++t) acc[t] = v8f_zero();

  for (int kb = 0; kb < 256; kb += 32) {
    // A operand: x rows fp32 -> bf16, CDNA5 16-bit A layout
    const float* xr = x + q * 256 + kb + half * 8;
    float4 f0 = *reinterpret_cast<const float4*>(xr);
    float4 f1 = *reinterpret_cast<const float4*>(xr + 4);
    float4 f2 = *reinterpret_cast<const float4*>(xr + 16);
    float4 f3 = *reinterpret_cast<const float4*>(xr + 20);
    v16bf a;
    a[0]=(bf16_t)f0.x; a[1]=(bf16_t)f0.y; a[2]=(bf16_t)f0.z; a[3]=(bf16_t)f0.w;
    a[4]=(bf16_t)f1.x; a[5]=(bf16_t)f1.y; a[6]=(bf16_t)f1.z; a[7]=(bf16_t)f1.w;
    a[8]=(bf16_t)f2.x; a[9]=(bf16_t)f2.y; a[10]=(bf16_t)f2.z; a[11]=(bf16_t)f2.w;
    a[12]=(bf16_t)f3.x; a[13]=(bf16_t)f3.y; a[14]=(bf16_t)f3.z; a[15]=(bf16_t)f3.w;
#pragma unroll
    for (int g = 0; g < 3; ++g) {
      v16bf bt[4];
#pragma unroll
      for (int j = 0; j < 4; ++j)
        bt[j] = load8x2(WcatT + ((g * 4 + j) * 16 + l16) * 256 + kb + half * 8, WcatT + ((g * 4 + j) * 16 + l16) * 256 + kb + 16 + half * 8);
#pragma unroll
      for (int j = 0; j < 4; ++j)
        acc[g * 4 + j] = wmma_bf16(a, bt[j], acc[g * 4 + j]);
    }
  }

#pragma unroll
  for (int t = 0; t < 12; ++t) {
    int cc = t * 16 + l16;
    float bias; bf16_t* dst; int col; int cols;
    if (t < 2)      { bias = bth[cc];      dst = theta; col = cc;      cols = 32; }
    else if (t < 4) { bias = bph[cc - 32]; dst = phiF;  col = cc - 32; cols = 32; }
    else            { bias = bg[cc - 64];  dst = gF;    col = cc - 64; cols = 128; }
#pragma unroll
    for (int r = 0; r < 8; ++r) {
      int m = r + half * 8;
      dst[(qb + m) * cols + col] = (bf16_t)(acc[t][r] + bias);
    }
  }
}

// ============================================================
// Kernel 2: 2x2 maxpool. phiP[b][key][32]; gT[b][vch][key] (transposed)
// ============================================================
__global__ __launch_bounds__(256) void pool2(
    const bf16_t* __restrict__ phiF, const bf16_t* __restrict__ gF,
    bf16_t* __restrict__ phiP, bf16_t* __restrict__ gT) {
  long id = (long)blockIdx.x * 256 + threadIdx.x;
  const long NPHI = 16L * 1024 * 32;
  if (id < NPHI) {
    int ch = (int)(id & 31);
    long t = id >> 5;
    int key = (int)(t & 1023);
    int b = (int)(t >> 10);
    int h2 = key >> 5, w2 = key & 31;
    float m = -3.4e38f;
#pragma unroll
    for (int dh = 0; dh < 2; ++dh)
#pragma unroll
      for (int dw = 0; dw < 2; ++dw) {
        long n = ((long)b * 64 + 2 * h2 + dh) * 64 + 2 * w2 + dw;
        m = fmaxf(m, (float)phiF[n * 32 + ch]);
      }
    phiP[id] = (bf16_t)m;
  } else {
    long j = id - NPHI;
    if (j < 16L * 128 * 1024) {
      int key = (int)(j & 1023);
      long t = j >> 10;
      int vch = (int)(t & 127);
      int b = (int)(t >> 7);
      int h2 = key >> 5, w2 = key & 31;
      float m = -3.4e38f;
#pragma unroll
      for (int dh = 0; dh < 2; ++dh)
#pragma unroll
        for (int dw = 0; dw < 2; ++dw) {
          long n = ((long)b * 64 + 2 * h2 + dh) * 64 + 2 * w2 + dw;
          m = fmaxf(m, (float)gF[n * 128 + vch]);
        }
      gT[j] = (bf16_t)m;
    }
  }
}

// ============================================================
// Kernel 3: flash attention + output conv + residual.
// WG = 8 waves, 128 queries; wave owns 16 queries.
// K/V chunk (32 keys) cooperatively staged in LDS once per WG;
// next chunk's global loads are in flight during compute
// (register double-buffering, explicit split barriers).
// ============================================================
__global__ __launch_bounds__(256) void attn_out(
    const bf16_t* __restrict__ theta, const bf16_t* __restrict__ phiP,
    const bf16_t* __restrict__ gT, const bf16_t* __restrict__ WoT,
    const float* __restrict__ battn, const float* __restrict__ sigma_p,
    const float* __restrict__ x, float* __restrict__ out) {
  __shared__ bf16_t Kbuf[32][32];    // keys x ch      (2 KB)
  __shared__ bf16_t Vbuf[128][32];   // vch  x keys    (8 KB)
  __shared__ bf16_t Pb[8][16][32];   // per-wave P relayout (8 KB)
  __shared__ bf16_t Ob[8][16][128];  // per-wave O relayout (32 KB)

  const int tid  = threadIdx.x;
  const int lane = tid & 31;
  const int wave = tid >> 5;
  const int half = lane >> 4;
  const int l16  = lane & 15;
  const int b  = blockIdx.x >> 5;   // 32 query tiles per batch
  const int qt = blockIdx.x & 31;
  const long q0 = (long)b * 4096 + qt * 128 + wave * 16;

  const bf16_t* phiB = phiP + (long)b * 1024 * 32;
  const bf16_t* gB   = gT   + (long)b * 128 * 1024;

  // cooperative-loader indices
  const int lkey = tid >> 3, lseg = tid & 7;  // phi: 32 keys x 8 segs of 4 bf16
  const int lvch = tid >> 1, lkh  = tid & 1;  // g: 128 vch x 2 halves of 16 bf16

  // Q A-operand (16 queries x 32 ch), resident whole kernel
  const bf16_t* th = theta + (q0 + l16) * 32;
  v16bf Qa = load8x2(th + half * 8, th + 16 + half * 8);

  float mrow[8], lrow[8];
  v8f acc[8];
#pragma unroll
  for (int r = 0; r < 8; ++r) { mrow[r] = -__builtin_inff(); lrow[r] = 0.0f; }
#pragma unroll
  for (int t = 0; t < 8; ++t) acc[t] = v8f_zero();
  const v8f zc = v8f_zero();

  // ---- prologue: stage chunk 0 ----
  uint2 kreg = *reinterpret_cast<const uint2*>(phiB + lkey * 32 + lseg * 4);
  uint4 vreg0 = *reinterpret_cast<const uint4*>(gB + lvch * 1024 + lkh * 16);
  uint4 vreg1 = *reinterpret_cast<const uint4*>(gB + lvch * 1024 + lkh * 16 + 8);
  *reinterpret_cast<uint2*>(&Kbuf[lkey][lseg * 4]) = kreg;
  *reinterpret_cast<uint4*>(&Vbuf[lvch][lkh * 16]) = vreg0;
  *reinterpret_cast<uint4*>(&Vbuf[lvch][lkh * 16 + 8]) = vreg1;
  asm volatile("s_wait_dscnt 0" ::: "memory");
  __builtin_amdgcn_s_barrier();

  for (int kt = 0; kt < 1024; kt += 32) {
    const bool hasNext = (kt + 32) < 1024;
    // issue next chunk's global loads now; waited only at the LDS-store point
    if (hasNext) {
      const int ktn = kt + 32;
      kreg  = *reinterpret_cast<const uint2*>(phiB + (ktn + lkey) * 32 + lseg * 4);
      vreg0 = *reinterpret_cast<const uint4*>(gB + lvch * 1024 + ktn + lkh * 16);
      vreg1 = *reinterpret_cast<const uint4*>(gB + lvch * 1024 + ktn + lkh * 16 + 8);
      __builtin_prefetch(gB + lvch * 1024 + ktn + 32, 0, 0);  // next-next V stream
    }

    // ---- S = Q * K^T for this 32-key chunk (B operands from LDS) ----
    v16bf K0 = load8x2(&Kbuf[l16][half * 8], &Kbuf[l16][16 + half * 8]);
    v16bf K1 = load8x2(&Kbuf[16 + l16][half * 8], &Kbuf[16 + l16][16 + half * 8]);
    v8f S0 = wmma_bf16(Qa, K0, zc);
    v8f S1 = wmma_bf16(Qa, K1, zc);

    // ---- online softmax: row max across 16-lane half ----
    float sc[8];
#pragma unroll
    for (int r = 0; r < 8; ++r) {
      float mx = fmaxf(S0[r], S1[r]);
      mx = fmaxf(mx, __shfl_xor(mx, 1, 32));
      mx = fmaxf(mx, __shfl_xor(mx, 2, 32));
      mx = fmaxf(mx, __shfl_xor(mx, 4, 32));
      mx = fmaxf(mx, __shfl_xor(mx, 8, 32));
      float nm = fmaxf(mrow[r], mx);
      sc[r] = __expf(mrow[r] - nm);
      mrow[r] = nm;
    }
    // ---- P = exp(S - m), row sums, stage P for A-operand relayout ----
#pragma unroll
    for (int r = 0; r < 8; ++r) {
      float e0 = __expf(S0[r] - mrow[r]);
      float e1 = __expf(S1[r] - mrow[r]);
      float s = e0 + e1;
      s += __shfl_xor(s, 1, 32);
      s += __shfl_xor(s, 2, 32);
      s += __shfl_xor(s, 4, 32);
      s += __shfl_xor(s, 8, 32);
      lrow[r] = lrow[r] * sc[r] + s;
      Pb[wave][r + half * 8][l16]      = (bf16_t)e0;
      Pb[wave][r + half * 8][16 + l16] = (bf16_t)e1;
    }
    asm volatile("s_wait_dscnt 0" ::: "memory");  // same-wave LDS RAW
    v16bf Pa = load8x2(&Pb[wave][l16][half * 8], &Pb[wave][l16][16 + half * 8]);

    // ---- rescale accumulators, then O += P * V (V from LDS) ----
#pragma unroll
    for (int t = 0; t < 8; ++t)
#pragma unroll
      for (int r = 0; r < 8; ++r) acc[t][r] = acc[t][r] * sc[r];
#pragma unroll
    for (int t = 0; t < 8; ++t) {
      v16bf Vb = load8x2(&Vbuf[t * 16 + l16][half * 8], &Vbuf[t * 16 + l16][16 + half * 8]);
      acc[t] = wmma_bf16(Pa, Vb, acc[t]);
    }

    // ---- rotate staging buffer ----
    if (hasNext) {
      asm volatile("s_wait_dscnt 0" ::: "memory");  // my LDS reads done
      __builtin_amdgcn_s_barrier();                 // everyone done reading
      *reinterpret_cast<uint2*>(&Kbuf[lkey][lseg * 4]) = kreg;    // waits loadcnt here
      *reinterpret_cast<uint4*>(&Vbuf[lvch][lkh * 16]) = vreg0;
      *reinterpret_cast<uint4*>(&Vbuf[lvch][lkh * 16 + 8]) = vreg1;
      asm volatile("s_wait_dscnt 0" ::: "memory");  // my stores visible
      __builtin_amdgcn_s_barrier();                 // everyone's stores visible
    }
  }

  // ---- finalize softmax: O /= l ; stage bf16 O tile in LDS ----
#pragma unroll
  for (int r = 0; r < 8; ++r) lrow[r] = 1.0f / lrow[r];
#pragma unroll
  for (int t = 0; t < 8; ++t)
#pragma unroll
    for (int r = 0; r < 8; ++r)
      Ob[wave][r + half * 8][t * 16 + l16] = (bf16_t)(acc[t][r] * lrow[r]);
  asm volatile("s_wait_dscnt 0" ::: "memory");

  // ---- output 1x1 conv: [16x128] x [128x256] ----
  v8f oc[16];
#pragma unroll
  for (int t = 0; t < 16; ++t) oc[t] = v8f_zero();
  for (int kb = 0; kb < 128; kb += 32) {
    v16bf Aa = load8x2(&Ob[wave][l16][kb + half * 8], &Ob[wave][l16][kb + 16 + half * 8]);
#pragma unroll
    for (int g = 0; g < 4; ++g) {
      v16bf bt[4];
#pragma unroll
      for (int j = 0; j < 4; ++j)
        bt[j] = load8x2(WoT + ((g * 4 + j) * 16 + l16) * 128 + kb + half * 8, WoT + ((g * 4 + j) * 16 + l16) * 128 + kb + 16 + half * 8);
#pragma unroll
      for (int j = 0; j < 4; ++j)
        oc[g * 4 + j] = wmma_bf16(Aa, bt[j], oc[g * 4 + j]);
    }
  }

  // ---- residual: out = x + sigma * (conv + bias) ----
  const float sg = sigma_p[0];
#pragma unroll
  for (int t = 0; t < 16; ++t) {
    int cout = t * 16 + l16;
    float bias = battn[cout];
#pragma unroll
    for (int r = 0; r < 8; ++r) {
      long idx = (q0 + r + half * 8) * 256 + cout;
      out[idx] = x[idx] + sg * (oc[t][r] + bias);
    }
  }
}

// ============================================================
extern "C" void kernel_launch(void* const* d_in, const int* in_sizes, int n_in,
                              void* d_out, int out_size, void* d_ws, size_t ws_size,
                              hipStream_t stream) {
  const float* x    = (const float*)d_in[0];
  const float* Wt   = (const float*)d_in[1];
  const float* bth  = (const float*)d_in[2];
  const float* Wp   = (const float*)d_in[3];
  const float* bph  = (const float*)d_in[4];
  const float* Wg   = (const float*)d_in[5];
  const float* bg   = (const float*)d_in[6];
  const float* Wo   = (const float*)d_in[7];
  const float* bat  = (const float*)d_in[8];
  const float* sig  = (const float*)d_in[9];
  float* out = (float*)d_out;

  char* ws = (char*)d_ws;
  bf16_t* WcatT = (bf16_t*)(ws + OFF_WCAT);
  bf16_t* WoT   = (bf16_t*)(ws + OFF_WOT);
  bf16_t* theta = (bf16_t*)(ws + OFF_THETA);
  bf16_t* phiF  = (bf16_t*)(ws + OFF_PHIF);
  bf16_t* gF    = (bf16_t*)(ws + OFF_GF);
  bf16_t* phiP  = (bf16_t*)(ws + OFF_PHIP);
  bf16_t* gTs   = (bf16_t*)(ws + OFF_GT);

  prep_w<<<(192 * 256 + 256 * 128 + 255) / 256, 256, 0, stream>>>(Wt, Wp, Wg, Wo, WcatT, WoT);
  conv_qkv<<<512, 256, 0, stream>>>(x, WcatT, bth, bph, bg, theta, phiF, gF);
  {
    long n = 16L * 1024 * 32 + 16L * 128 * 1024;
    pool2<<<(int)((n + 255) / 256), 256, 0, stream>>>(phiF, gF, phiP, gTs);
  }
  attn_out<<<512, 256, 0, stream>>>(theta, phiP, gTs, WoT, bat, sig, x, out);
}
